// GCN_experimental_41455024341475
// MI455X (gfx1250) — matrix operand order repaired and checked by reading, not compiled, not run on hardware
//
#include <hip/hip_runtime.h>
#include <hip/hip_bf16.h>

// ---------------------------------------------------------------------------
// GCN (5-layer) on MI455X / gfx1250.
//  * Dense GEMMs on v_wmma_f32_16x16x32_f16 (fp16 A/B, fp32 accumulate).
//  * All WMMA fragment loads are UNGUARDED vector loads (row clamp + zero
//    padding instead of predication) -> no divergent load/wait chains.
//  * Edge aggregation = fp32 hardware atomics into L2-resident buffers
//    (working set << 192MB L2; HBM sees ~one streaming pass over x + edges).
//  * Everything launched on `stream`, no sync / malloc -> graph-capturable.
// ---------------------------------------------------------------------------

typedef __attribute__((ext_vector_type(16))) _Float16 v16h;
typedef __attribute__((ext_vector_type(8)))  _Float16 v8h;
typedef __attribute__((ext_vector_type(8)))  float    v8f;
typedef __attribute__((ext_vector_type(4)))  float    v4f;

__device__ __forceinline__ void atomAddF(float* p, float v) {
  unsafeAtomicAdd(p, v);   // hardware global_atomic_add_f32
}

__device__ __forceinline__ int imin(int a, int b) { return a < b ? a : b; }

// ---------------------------------------------------------------------------
// Pack W[K,Cout] (f32) -> WhT[16][Kp] (f16, transposed, zero-padded).
// B fragment then = 16 contiguous halves at WhT[col*Kp + k0 + hi*16].
// ---------------------------------------------------------------------------
__global__ void k_pack_w(const float* __restrict__ W, _Float16* __restrict__ WhT,
                         int K, int Cout, int Kp) {
  int t = blockIdx.x * blockDim.x + threadIdx.x;
  if (t < 16 * Kp) {
    const int col = t / Kp, k = t % Kp;
    const float v = (k < K && col < Cout) ? W[k * Cout + col] : 0.0f;
    WhT[col * Kp + k] = (_Float16)v;
  }
}

// ---------------------------------------------------------------------------
// GEMM, A in f32 (layer 1): out[M,Cout] = A[M,K] * W[K,Cout].
// Requires K % 32 == 0 (true for F=128). One 16-row tile per wave32.
// A fragment: per lane, floats A[row][k0+hi*16 .. +15].
// ---------------------------------------------------------------------------
__global__ void gemm_wmma_f32A(const float* __restrict__ A,
                               const _Float16* __restrict__ WhT,
                               float* __restrict__ out,
                               int M, int K, int Cout, int ntiles) {
  const int wave = blockIdx.x * (blockDim.x >> 5) + (threadIdx.x >> 5);
  if (wave >= ntiles) return;                       // wave-uniform
  const int lane   = threadIdx.x & 31;
  const int hi     = lane >> 4;
  const int lane16 = lane & 15;
  const int row    = imin(wave * 16 + lane16, M - 1);   // clamp, no predication
  const float* __restrict__ arow = A + (long)row * K;

  v8f c = {};
  for (int k0 = 0; k0 < K; k0 += 32) {
    const v4f f0 = *(const v4f*)(arow + k0 + hi * 16);
    const v4f f1 = *(const v4f*)(arow + k0 + hi * 16 + 4);
    const v4f f2 = *(const v4f*)(arow + k0 + hi * 16 + 8);
    const v4f f3 = *(const v4f*)(arow + k0 + hi * 16 + 12);
    v16h a;
#pragma unroll
    for (int e = 0; e < 4; ++e) {
      a[e]      = (_Float16)f0[e];
      a[e + 4]  = (_Float16)f1[e];
      a[e + 8]  = (_Float16)f2[e];
      a[e + 12] = (_Float16)f3[e];
    }
    const v16h b = *(const v16h*)(WhT + lane16 * K + k0 + hi * 16);  // Kp == K here
    c = __builtin_amdgcn_wmma_f32_16x16x32_f16(false, a, false, b, (short)0, c,
                                               false, false);
  }
  if (lane16 < Cout) {
#pragma unroll
    for (int j = 0; j < 8; ++j) {
      const int r = wave * 16 + j + hi * 8;
      if (r < M) out[(long)r * Cout + lane16] = c[j];
    }
  }
}

// ---------------------------------------------------------------------------
// GEMM, A in f16 padded stride 32 (layers 2..5): K <= 32, single WMMA.
// Ah[M][32] zero-padded; WhT[16][32] zero-padded.
// ---------------------------------------------------------------------------
__global__ void gemm_wmma_f16A(const _Float16* __restrict__ Ah,
                               const _Float16* __restrict__ WhT,
                               float* __restrict__ out,
                               int M, int Cout, int ntiles) {
  const int wave = blockIdx.x * (blockDim.x >> 5) + (threadIdx.x >> 5);
  if (wave >= ntiles) return;
  const int lane   = threadIdx.x & 31;
  const int hi     = lane >> 4;
  const int lane16 = lane & 15;
  const int row    = imin(wave * 16 + lane16, M - 1);
  const _Float16* __restrict__ arow = Ah + (long)row * 32;

  const v8h lo = *(const v8h*)(arow + hi * 16);
  const v8h hh = *(const v8h*)(arow + hi * 16 + 8);
  v16h a;
#pragma unroll
  for (int e = 0; e < 8; ++e) { a[e] = lo[e]; a[e + 8] = hh[e]; }
  const v16h b = *(const v16h*)(WhT + lane16 * 32 + hi * 16);

  v8f c = {};
  c = __builtin_amdgcn_wmma_f32_16x16x32_f16(false, a, false, b, (short)0, c,
                                             false, false);
  if (lane16 < Cout) {
#pragma unroll
    for (int j = 0; j < 8; ++j) {
      const int r = wave * 16 + j + hi * 8;
      if (r < M) out[(long)r * Cout + lane16] = c[j];
    }
  }
}

// ---------------------------------------------------------------------------
// Degree / normalization
// ---------------------------------------------------------------------------
__global__ void k_deg_init(float* __restrict__ deg, int N) {
  int i = blockIdx.x * blockDim.x + threadIdx.x;
  if (i < N) deg[i] = 1.0f;                  // self-loop
}
__global__ void k_deg_edges(const int* __restrict__ dst, float* __restrict__ deg, int E) {
  int e = blockIdx.x * blockDim.x + threadIdx.x;
  if (e < E) atomAddF(&deg[dst[e]], 1.0f);
}
__global__ void k_dinv(float* __restrict__ d, int N) {
  int i = blockIdx.x * blockDim.x + threadIdx.x;
  if (i < N) d[i] = rsqrtf(d[i]);            // deg >= 1 always
}

// ---------------------------------------------------------------------------
// Aggregation: agg[d] = dinv[d]^2*h[d] (self-loop init) + sum_e dinv[s]dinv[d]h[s]
// ---------------------------------------------------------------------------
template <int C>
__global__ void k_agg_init(const float* __restrict__ h, const float* __restrict__ dinv,
                           float* __restrict__ agg, int N) {
  int i = blockIdx.x * blockDim.x + threadIdx.x;
  if (i < N) {
    float s = dinv[i]; s *= s;
#pragma unroll
    for (int c = 0; c < C; ++c) agg[(long)i * C + c] = s * h[(long)i * C + c];
  }
}

template <int C>
__global__ void k_agg_edges(const int* __restrict__ src, const int* __restrict__ dst,
                            const float* __restrict__ dinv, const float* __restrict__ h,
                            float* __restrict__ agg, int E) {
  int e = blockIdx.x * blockDim.x + threadIdx.x;
  if (e < E) {
    const int s = src[e], d = dst[e];
    const float n = dinv[s] * dinv[d];
#pragma unroll
    for (int c = 0; c < C; ++c)
      atomAddF(&agg[(long)d * C + c], n * h[(long)s * C + c]);
  }
}

// bias + ReLU -> f16 activation, padded stride 32 (feeds next WMMA GEMM)
template <int C>
__global__ void k_bias_relu_h(const float* __restrict__ agg, const float* __restrict__ b,
                              _Float16* __restrict__ Ah, int N) {
  int i = blockIdx.x * blockDim.x + threadIdx.x;
  if (i < N) {
    v16h w0 = {}, w1 = {};
#pragma unroll
    for (int c = 0; c < C; ++c) {
      float v = agg[(long)i * C + c] + b[c];
      v = v > 0.0f ? v : 0.0f;
      if (c < 16) w0[c] = (_Float16)v; else w1[c - 16] = (_Float16)v;
    }
    *(v16h*)(Ah + (long)i * 32)      = w0;
    *(v16h*)(Ah + (long)i * 32 + 16) = w1;
  }
}

// bias + ReLU -> f32 dense (last layer, feeds pooling)
template <int C>
__global__ void k_bias_relu_f(const float* __restrict__ agg, const float* __restrict__ b,
                              float* __restrict__ outp, int N) {
  int t = blockIdx.x * blockDim.x + threadIdx.x;
  if (t < N * C) {
    const float v = agg[t] + b[t % C];
    outp[t] = v > 0.0f ? v : 0.0f;
  }
}

// ---------------------------------------------------------------------------
// Mean pool + log_softmax
// ---------------------------------------------------------------------------
template <int C>
__global__ void k_pool_zero(float* __restrict__ sums, float* __restrict__ cnts, int G) {
  int t = blockIdx.x * blockDim.x + threadIdx.x;
  if (t < G * C) sums[t] = 0.0f;
  if (t < G)     cnts[t] = 0.0f;
}

template <int C>
__global__ void k_pool_acc(const float* __restrict__ h, const int* __restrict__ batch,
                           float* __restrict__ sums, float* __restrict__ cnts, int N) {
  int i = blockIdx.x * blockDim.x + threadIdx.x;
  if (i < N) {
    const int g = batch[i];
#pragma unroll
    for (int c = 0; c < C; ++c) atomAddF(&sums[(long)g * C + c], h[(long)i * C + c]);
    atomAddF(&cnts[g], 1.0f);
  }
}

template <int C>
__global__ void k_logsoftmax(const float* __restrict__ sums, const float* __restrict__ cnts,
                             float* __restrict__ out, int G) {
  int g = blockIdx.x * blockDim.x + threadIdx.x;
  if (g < G) {
    float cnt = cnts[g]; cnt = cnt > 1.0f ? cnt : 1.0f;
    float p[C], m = -3.4e38f;
#pragma unroll
    for (int c = 0; c < C; ++c) { p[c] = sums[g * C + c] / cnt; m = p[c] > m ? p[c] : m; }
    float s = 0.0f;
#pragma unroll
    for (int c = 0; c < C; ++c) s += expf(p[c] - m);
    const float l = logf(s) + m;
#pragma unroll
    for (int c = 0; c < C; ++c) out[g * C + c] = p[c] - l;
  }
}

// ---------------------------------------------------------------------------
// Host-side orchestration (all on `stream`, graph-capture safe)
// ---------------------------------------------------------------------------
extern "C" void kernel_launch(void* const* d_in, const int* in_sizes, int n_in,
                              void* d_out, int out_size, void* d_ws, size_t ws_size,
                              hipStream_t stream) {
  const float* x     = (const float*)d_in[0];
  const int*   ei    = (const int*)  d_in[1];
  const int*   batch = (const int*)  d_in[2];
  const float* W1 = (const float*)d_in[3];  const float* b1 = (const float*)d_in[4];
  const float* W2 = (const float*)d_in[5];  const float* b2 = (const float*)d_in[6];
  const float* W3 = (const float*)d_in[7];  const float* b3 = (const float*)d_in[8];
  const float* W4 = (const float*)d_in[9];  const float* b4 = (const float*)d_in[10];
  const float* Wf = (const float*)d_in[11]; const float* bf = (const float*)d_in[12];

  const int N = in_sizes[2];
  const int E = in_sizes[1] / 2;
  constexpr int H = 6;
  constexpr int C = 10;
  const int F = in_sizes[3] / H;        // 128 (multiple of 32)
  const int G = out_size / C;           // 512

  const int* src = ei;
  const int* dst = ei + E;

  // workspace (floats): dinv[N] | lin[N*C] | agg[N*C] | Ah[N*32 f16 = N*16 f32]
  //                     | sums[G*C] | cnts[G] | WhT[16*128 f16 = 1024 f32]
  float* ws   = (float*)d_ws;
  float* dinv = ws;
  float* lin  = dinv + N;
  float* agg  = lin  + (size_t)N * C;
  _Float16* Ah = (_Float16*)(agg + (size_t)N * C);
  float* sums = (float*)(Ah + (size_t)N * 32);
  float* cnts = sums + (size_t)G * C;
  _Float16* WhT = (_Float16*)(cnts + G);

  auto cdiv = [](int a, int b) { return (a + b - 1) / b; };
  const int BT = 256;
  const int ntiles = cdiv(N, 16);
  const dim3 gGemm(cdiv(ntiles, 8));    // 8 waves (one 16-row tile each) per block

  // --- normalization ---
  k_deg_init <<<cdiv(N, BT), BT, 0, stream>>>(dinv, N);
  k_deg_edges<<<cdiv(E, BT), BT, 0, stream>>>(dst, dinv, E);
  k_dinv     <<<cdiv(N, BT), BT, 0, stream>>>(dinv, N);

  // --- layer 1: [N,F] x [F,H], f32 A path ---
  k_pack_w<<<cdiv(16 * F, BT), BT, 0, stream>>>(W1, WhT, F, H, F);
  gemm_wmma_f32A<<<gGemm, 256, 0, stream>>>(x, WhT, lin, N, F, H, ntiles);
  k_agg_init<H> <<<cdiv(N, BT), BT, 0, stream>>>(lin, dinv, agg, N);
  k_agg_edges<H><<<cdiv(E, BT), BT, 0, stream>>>(src, dst, dinv, lin, agg, E);
  k_bias_relu_h<H><<<cdiv(N, BT), BT, 0, stream>>>(agg, b1, Ah, N);

  // --- layers 2..4: [N,H] x [H,H], f16 A path ---
  const float* Ws[3] = {W2, W3, W4};
  const float* bs[3] = {b2, b3, b4};
  for (int l = 0; l < 3; ++l) {
    k_pack_w<<<cdiv(16 * 32, BT), BT, 0, stream>>>(Ws[l], WhT, H, H, 32);
    gemm_wmma_f16A<<<gGemm, 256, 0, stream>>>(Ah, WhT, lin, N, H, ntiles);
    k_agg_init<H> <<<cdiv(N, BT), BT, 0, stream>>>(lin, dinv, agg, N);
    k_agg_edges<H><<<cdiv(E, BT), BT, 0, stream>>>(src, dst, dinv, lin, agg, E);
    k_bias_relu_h<H><<<cdiv(N, BT), BT, 0, stream>>>(agg, bs[l], Ah, N);
  }

  // --- layer 5: [N,H] x [H,C], f16 A path, f32 output for pooling ---
  k_pack_w<<<cdiv(16 * 32, BT), BT, 0, stream>>>(Wf, WhT, H, C, 32);
  gemm_wmma_f16A<<<gGemm, 256, 0, stream>>>(Ah, WhT, lin, N, C, ntiles);
  k_agg_init<C> <<<cdiv(N, BT), BT, 0, stream>>>(lin, dinv, agg, N);
  k_agg_edges<C><<<cdiv(E, BT), BT, 0, stream>>>(src, dst, dinv, lin, agg, E);
  k_bias_relu_f<C><<<cdiv(N * C, BT), BT, 0, stream>>>(agg, bf, lin, N);

  // --- mean pool + log_softmax ---
  k_pool_zero<C><<<cdiv(G * C, BT), BT, 0, stream>>>(sums, cnts, G);
  k_pool_acc<C> <<<cdiv(N, BT), BT, 0, stream>>>(lin, batch, sums, cnts, N);
  k_logsoftmax<C><<<cdiv(G, BT), BT, 0, stream>>>(sums, cnts, (float*)d_out, G);
}
